// CMLILoss_51359218925721
// MI455X (gfx1250) — matrix operand order repaired and checked by reading, not compiled, not run on hardware
//
#include <hip/hip_runtime.h>

typedef __attribute__((ext_vector_type(16))) _Float16 v16h;
typedef __attribute__((ext_vector_type(8)))  float    v8f;
typedef unsigned int u32x4 __attribute__((ext_vector_type(4)));
typedef int          i32x8 __attribute__((ext_vector_type(8)));
typedef int          i32x4 __attribute__((ext_vector_type(4)));

#define Bsz 64
#define T_  128
#define I_  197
#define D_  768
#define NPAD 224   // image tokens padded 196 -> 224
#define NBLK 112   // N tile per block (2 blocks cover 224)
#define KC   64    // K chunk staged in LDS
#define NC   (D_ / KC)

#if defined(__has_builtin)
#  if __has_builtin(__builtin_amdgcn_tensor_load_to_lds)
#    define USE_TDM 1
#  else
#    define USE_TDM 0
#  endif
#else
#  define USE_TDM 0
#endif

union Frag { uint4 q[2]; v16h h; };

// ---------------- EOS mask fixup: mark last zero of padding_mask as padded --
__global__ void k_eos(const int* __restrict__ pm, int* __restrict__ epad) {
    int b = blockIdx.x; int t = threadIdx.x;           // 128 threads
    __shared__ int s_lz[T_];
    int v = pm[b * T_ + t];
    s_lz[t] = (v == 0) ? t : -1;
    __syncthreads();
    for (int o = 64; o > 0; o >>= 1) {
        if (t < o) s_lz[t] = max(s_lz[t], s_lz[t + o]);
        __syncthreads();
    }
    int lz = s_lz[0];
    epad[b * T_ + t] = (v != 0 || t == lz) ? 1 : 0;
}

// ---------------- fp32 -> fp16 conversions into padded layouts -------------
__global__ void k_cvt_txt(const float* __restrict__ txt, _Float16* __restrict__ out) {
    long idx = (long)blockIdx.x * blockDim.x + threadIdx.x;   // [64][128][768]
    int d = (int)(idx % D_);
    long t = idx / D_;
    int r = (int)(t % T_);
    int b = (int)(t / T_);
    float v = (r < T_ - 1) ? txt[((long)b * T_ + r + 1) * D_ + d] : 0.f;
    out[idx] = (_Float16)v;
}

__global__ void k_cvt_img(const float* __restrict__ img, _Float16* __restrict__ out) {
    long idx = (long)blockIdx.x * blockDim.x + threadIdx.x;   // [64][224][768]
    int d = (int)(idx % D_);
    long t = idx / D_;
    int r = (int)(t % NPAD);
    int b = (int)(t / NPAD);
    float v = (r < I_ - 1) ? img[((long)b * I_ + r + 1) * D_ + d] : 0.f;
    out[idx] = (_Float16)v;
}

// ---------------- shared memory: tiles overlaid with epilogue buffers ------
struct TilesT {
    _Float16 A[2][T_][KC];      // 2 x 16 KB
    _Float16 B[2][NBLK][KC];    // 2 x 14 KB
};
struct RedT {
    float colbuf[8][NBLK];
    int   padv[T_];
};
union SMemT {
    TilesT tiles;
    RedT   red;
};

#if USE_TDM
// Issue a 2D TDM tile load: tile_w (elems, contiguous dim) x tile_h rows from a
// row-major fp16 matrix with row stride `rstride` elems, into LDS at lds_addr.
__device__ __forceinline__ void tdm_load_2d(unsigned lds_addr, const _Float16* gptr,
                                            unsigned tile_w, unsigned tile_h,
                                            unsigned rstride, unsigned rows) {
    unsigned long long ga = (unsigned long long)(size_t)gptr;
    u32x4 g0;
    g0[0] = 1u;                                              // count=1, user mode
    g0[1] = lds_addr;                                        // lds_addr [63:32]
    g0[2] = (unsigned)ga;                                    // global_addr [95:64]
    g0[3] = (unsigned)((ga >> 32) & 0x01FFFFFFu) | (2u << 30); // addr[56:32] | type=2
    i32x8 g1;
    unsigned td0 = rstride;                                  // tensor_dim0 (elems)
    unsigned td1 = rows;                                     // tensor_dim1
    g1[0] = (int)(1u << 16);                                 // data_size=1 -> 2 bytes
    g1[1] = (int)((td0 & 0xFFFFu) << 16);                    // tensor_dim0[15:0]
    g1[2] = (int)((td0 >> 16) | ((td1 & 0xFFFFu) << 16));    // dim0[31:16]|dim1[15:0]
    g1[3] = (int)((td1 >> 16) | (tile_w << 16));             // dim1[31:16]|tile_dim0
    g1[4] = (int)tile_h;                                     // tile_dim1, tile_dim2=0
    g1[5] = (int)rstride;                                    // tensor_dim0_stride lo
    g1[6] = 0;                                               // stride0 hi | stride1 lo
    g1[7] = 0;                                               // stride1 hi
    i32x4 z4 = {0, 0, 0, 0};                                 // groups 2/3 unused (2D)
    i32x8 z8 = {0, 0, 0, 0, 0, 0, 0, 0};
    __builtin_amdgcn_tensor_load_to_lds(g0, g1, z4, z4, z8, 0);
}
#endif

// ---------------- fused batched GEMM + masked max reductions ---------------
// grid = (2, 64, 64): (N-half, image batch y, text batch x); block = 256 (8 waves)
__global__ __launch_bounds__(256) void k_cmli_gemm(
    const _Float16* __restrict__ txt16, const _Float16* __restrict__ img16,
    const int* __restrict__ epad, const float* __restrict__ logit_scale,
    float* __restrict__ t2i_part, float* __restrict__ colmax) {

    __shared__ alignas(16) SMemT smem;

    const int nb = blockIdx.x, y = blockIdx.y, x = blockIdx.z;
    const int tid = threadIdx.x;
    const int w = tid >> 5, lane = tid & 31;
    const int laneM = lane & 15, laneHalf = lane >> 4;

    const float scale = (float)(_Float16)logit_scale[0];   // reference casts to fp16

    v8f acc[7];
    const v8f vzero = {0.f, 0.f, 0.f, 0.f, 0.f, 0.f, 0.f, 0.f};
#pragma unroll
    for (int n = 0; n < 7; n++) acc[n] = vzero;

    const _Float16* aBase = txt16 + (size_t)x * T_ * D_;
    const _Float16* bBase = img16 + ((size_t)y * NPAD + nb * NBLK) * D_;

#if USE_TDM
    const unsigned ldsA0 = (unsigned)(size_t)&smem.tiles.A[0][0][0];
    const unsigned ldsA1 = (unsigned)(size_t)&smem.tiles.A[1][0][0];
    const unsigned ldsB0 = (unsigned)(size_t)&smem.tiles.B[0][0][0];
    const unsigned ldsB1 = (unsigned)(size_t)&smem.tiles.B[1][0][0];
    if (w == 0) {                                  // one wave drives the TDM
        tdm_load_2d(ldsA0, aBase, KC, T_, D_, T_);
        tdm_load_2d(ldsB0, bBase, KC, NBLK, D_, NBLK);
    }
#endif

    for (int c = 0; c < NC; ++c) {
        const int buf = c & 1;
#if USE_TDM
        if (w == 0) __builtin_amdgcn_s_wait_tensorcnt(0);  // chunk c landed in LDS
        __syncthreads();                                   // ...visible to all waves
        if (w == 0 && c + 1 < NC) {                        // prefetch chunk c+1 into
            const int k1 = (c + 1) * KC;                   // the other buffer via TDM
            tdm_load_2d(buf ? ldsA0 : ldsA1, aBase + k1, KC, T_, D_, T_);
            tdm_load_2d(buf ? ldsB0 : ldsB1, bBase + k1, KC, NBLK, D_, NBLK);
        }
#else
        const int k0 = c * KC;
        __syncthreads();
#pragma unroll
        for (int i = 0; i < 4; i++) {                      // A: 1024 uint4
            int idx = tid + i * 256;
            int row = idx >> 3, q = idx & 7;
            ((uint4*)&smem.tiles.A[buf][0][0])[idx] =
                *(const uint4*)(aBase + (size_t)row * D_ + k0 + q * 8);
        }
#pragma unroll
        for (int i = 0; i < 4; i++) {                      // B: 896 uint4
            int idx = tid + i * 256;
            if (idx < 896) {
                int row = idx >> 3, q = idx & 7;
                ((uint4*)&smem.tiles.B[buf][0][0])[idx] =
                    *(const uint4*)(bBase + (size_t)row * D_ + k0 + q * 8);
            }
        }
        __syncthreads();
#endif

#pragma unroll
        for (int ks = 0; ks < KC; ks += 32) {
            // A fragment 16x32: lanes 0-15 take K ks+0..7 / ks+16..23,
            // lanes 16-31 take K ks+8..15 / ks+24..31  (two ds_load_b128)
            Frag af;
            const _Float16* ap = &smem.tiles.A[buf][w * 16 + laneM][ks + laneHalf * 8];
            af.q[0] = *(const uint4*)ap;
            af.q[1] = *(const uint4*)(ap + 16);
            // Preload all 7 B fragments, then run the WMMA chain back-to-back
            Frag bf[7];
#pragma unroll
            for (int n = 0; n < 7; n++) {
                const _Float16* bp =
                    &smem.tiles.B[buf][n * 16 + laneM][ks + laneHalf * 8];
                bf[n].q[0] = *(const uint4*)bp;
                bf[n].q[1] = *(const uint4*)(bp + 16);
            }
#pragma unroll
            for (int n = 0; n < 7; n++)
                acc[n] = __builtin_amdgcn_wmma_f32_16x16x32_f16(
                    false, af.h, false, bf[n].h, (short)0, acc[n], false, false);
        }
    }

    __syncthreads();   // all tile reads done; safe to overlay epilogue buffers
    if (tid < T_) smem.red.padv[tid] = epad[x * T_ + tid];

    // ---- row max over i (t2i_tok partial), masking padded i columns ----
    // C/D layout: VGPR v, lane l -> row = v + (l<16?0:8), col = l%16
    float rm[8];
#pragma unroll
    for (int v = 0; v < 8; v++) rm[v] = -3.4e38f;
#pragma unroll
    for (int n = 0; n < 7; n++) {
        int col_g = nb * NBLK + n * 16 + laneM;        // global image-token col
        bool cvalid = col_g < (I_ - 1);                // 196 real tokens
#pragma unroll
        for (int v = 0; v < 8; v++) {
            float s = scale * acc[n][v];
            if (cvalid) rm[v] = fmaxf(rm[v], s);
        }
    }
    for (int m = 8; m >= 1; m >>= 1) {
#pragma unroll
        for (int v = 0; v < 8; v++)
            rm[v] = fmaxf(rm[v], __shfl_xor(rm[v], m, 32));
    }
    if (laneM == 0) {
        size_t base = (((size_t)x * Bsz + y) * 2 + nb) * T_ + w * 16 + laneHalf * 8;
#pragma unroll
        for (int v = 0; v < 8; v++) t2i_part[base + v] = rm[v];
    }

    __syncthreads();   // padv visible

    // ---- col max over t (i2t partial), padded t rows -> -65504 (fp16 lowest)
#pragma unroll
    for (int n = 0; n < 7; n++) {
        float cm = -3.4e38f;
#pragma unroll
        for (int v = 0; v < 8; v++) {
            int t_row = w * 16 + laneHalf * 8 + v;     // 0..127 (row 127 = pad)
            bool rvalid = (t_row < T_ - 1) && (smem.red.padv[t_row + 1] == 0);
            float s = rvalid ? scale * acc[n][v] : -65504.0f;
            cm = fmaxf(cm, s);
        }
        cm = fmaxf(cm, __shfl_xor(cm, 16, 32));        // merge the two row halves
        if (lane < 16) smem.red.colbuf[w][n * 16 + lane] = cm;
    }
    __syncthreads();
    if (tid < NBLK) {
        float m = smem.red.colbuf[0][tid];
#pragma unroll
        for (int ww = 1; ww < 8; ww++) m = fmaxf(m, smem.red.colbuf[ww][tid]);
        colmax[((size_t)x * Bsz + y) * NPAD + nb * NBLK + tid] = m;
    }
}

// ---------------- per-(x,y) reductions -> i2t, t2i -------------------------
__global__ __launch_bounds__(256) void k_reduce_pair(
    const float* __restrict__ t2i_part, const float* __restrict__ colmax,
    const int* __restrict__ epad, float* __restrict__ out) {
    int y = blockIdx.x, x = blockIdx.y;
    int tid = threadIdx.x;
    __shared__ float s_a[256];
    __shared__ float s_b[256];

    // t2i: masked mean over t of max-over-i
    float numer = 0.f, cnt = 0.f;
    if (tid < T_) {
        size_t base = (((size_t)x * Bsz + y) * 2) * T_;
        float m = fmaxf(t2i_part[base + tid], t2i_part[base + T_ + tid]);
        bool valid = (tid < T_ - 1) && (epad[x * T_ + tid + 1] == 0);
        numer = valid ? m : 0.f;
        cnt   = valid ? 1.f : 0.f;
    }
    s_a[tid] = numer; s_b[tid] = cnt;
    __syncthreads();
    for (int o = 128; o > 0; o >>= 1) {
        if (tid < o) { s_a[tid] += s_a[tid + o]; s_b[tid] += s_b[tid + o]; }
        __syncthreads();
    }
    if (tid == 0) {
        float denom = fmaxf(s_b[0], 1e-6f);
        out[1 + 4096 + x * Bsz + y] = (float)(_Float16)(s_a[0] / denom);
    }
    __syncthreads();

    // i2t: mean over the 196 real image tokens of masked max-over-t
    float v = 0.f;
    if (tid < I_ - 1) v = colmax[((size_t)x * Bsz + y) * NPAD + tid];
    s_a[tid] = v;
    __syncthreads();
    for (int o = 128; o > 0; o >>= 1) {
        if (tid < o) s_a[tid] += s_a[tid + o];
        __syncthreads();
    }
    if (tid == 0)
        out[1 + x * Bsz + y] = (float)(_Float16)(s_a[0] / (float)(I_ - 1));
}

// ---------------- diagonal cross-entropy loss ------------------------------
__global__ __launch_bounds__(128) void k_loss(float* __restrict__ out) {
    int tid = threadIdx.x;
    int r = tid & 63, which = tid >> 6;       // 0: i2t, 1: t2i
    const float* M = out + 1 + which * 4096;
    float mx = -3.4e38f;
    for (int c = 0; c < Bsz; c++) mx = fmaxf(mx, M[r * Bsz + c]);
    float se = 0.f;
    for (int c = 0; c < Bsz; c++) se += expf(M[r * Bsz + c] - mx);
    float lp = M[r * Bsz + r] - mx - logf(se);
    __shared__ float s[128];
    s[tid] = lp;
    __syncthreads();
    for (int o = 64; o > 0; o >>= 1) {
        if (tid < o) s[tid] += s[tid + o];
        __syncthreads();
    }
    if (tid == 0) out[0] = -s[0] / 128.0f;    // (ce_i2t + ce_t2i)/2
}

extern "C" void kernel_launch(void* const* d_in, const int* in_sizes, int n_in,
                              void* d_out, int out_size, void* d_ws, size_t ws_size,
                              hipStream_t stream) {
    const float* img    = (const float*)d_in[0];   // (64,197,768) f32
    const float* txt    = (const float*)d_in[1];   // (64,128,768) f32
    const int*   pm     = (const int*)d_in[2];     // (64,128) i32
    const float* lscale = (const float*)d_in[3];   // (1,) f32
    float* out = (float*)d_out;                    // [loss | i2t 64x64 | t2i 64x64]

    char* ws = (char*)d_ws;
    int*      epad     = (int*)(ws + 0);                       //  32 KB
    _Float16* txt16    = (_Float16*)(ws + 32768);              //  12.58 MB
    _Float16* img16    = (_Float16*)(ws + 32768 + 12582912);   //  22.02 MB
    float*    t2i_part = (float*)(ws + 34635776);              //   4.19 MB
    float*    colmax   = (float*)(ws + 38830080);              //   3.67 MB

    k_eos<<<Bsz, T_, 0, stream>>>(pm, epad);
    k_cvt_txt<<<(Bsz * T_ * D_) / 256, 256, 0, stream>>>(txt, txt16);
    k_cvt_img<<<(Bsz * NPAD * D_) / 256, 256, 0, stream>>>(img, img16);
    k_cmli_gemm<<<dim3(2, Bsz, Bsz), 256, 0, stream>>>(txt16, img16, epad, lscale,
                                                       t2i_part, colmax);
    k_reduce_pair<<<dim3(Bsz, Bsz), 256, 0, stream>>>(t2i_part, colmax, epad, out);
    k_loss<<<1, 128, 0, stream>>>(out);
}
